// SlimmableMultiHeadAttention_8366596292691
// MI455X (gfx1250) — compile-verified, matrix operand order checked by reading, not run on hardware
//
#include <hip/hip_runtime.h>
#include <math.h>
#include <stdint.h>

typedef _Float16 v16h __attribute__((ext_vector_type(16)));
typedef _Float16 v8h  __attribute__((ext_vector_type(8)));
typedef _Float16 v4h  __attribute__((ext_vector_type(4)));
typedef float    v8f  __attribute__((ext_vector_type(8)));
typedef float    v4f  __attribute__((ext_vector_type(4)));

#define WMMA_F16(A, B, C) \
  __builtin_amdgcn_wmma_f32_16x16x32_f16(false, (A), false, (B), (short)0, (C), false, false)

// Load one 16-lane-striped f16 fragment row per the CDNA5 16-bit A/B layout:
// lane holds 16 K-values as two contiguous 8-half (16B) chunks at kb and kb+16.
__device__ __forceinline__ v16h frag_from(const _Float16* p, int kb) {
  v8h lo = *(const v8h*)(p + kb);
  v8h hi = *(const v8h*)(p + kb + 16);
  v16h r;
#pragma unroll
  for (int i = 0; i < 8; ++i) { r[i] = lo[i]; r[i + 8] = hi[i]; }
  return r;
}

__device__ __forceinline__ void wait_lds() {
#if __has_builtin(__builtin_amdgcn_s_wait_dscnt)
  __builtin_amdgcn_s_wait_dscnt(0);
#else
  asm volatile("s_wait_dscnt 0" ::: "memory");
#endif
}

// Async global->LDS 16-byte copy (ASYNCcnt-tracked). lds_off is the 32-bit
// LDS byte address (low 32 bits of the flat shared pointer).
__device__ __forceinline__ void async_b128(const _Float16* g, unsigned lds_off) {
  asm volatile("global_load_async_to_lds_b128 %0, %1, off"
               :: "v"(lds_off), "v"(g) : "memory");
}

#if __has_builtin(__builtin_amdgcn_s_wait_asynccnt)
#define WAIT_ASYNC(n) __builtin_amdgcn_s_wait_asynccnt(n)
#else
#define WAIT_ASYNC(n) asm volatile("s_wait_asynccnt %0" :: "i"(n) : "memory")
#endif

// ---------------------------------------------------------------------------
// f32 -> packed f16 conversion (strided source).
// ---------------------------------------------------------------------------
__global__ __launch_bounds__(256) void cvt_f16(const float* __restrict__ src,
                                               _Float16* __restrict__ dst,
                                               int cols, int lds_) {
  const int r = blockIdx.y;
  const int c = (blockIdx.x * 256 + threadIdx.x) * 4;
  if (c >= cols) return;
  v4f v = *(const v4f*)(src + (size_t)r * lds_ + c);
  v4h o;
#pragma unroll
  for (int e = 0; e < 4; ++e) o[e] = (_Float16)v[e];
  *(v4h*)(dst + (size_t)r * cols + c) = o;
}

// ---------------------------------------------------------------------------
// Tiled GEMM (all-f16 operands, f32 accum):
//   Y[m,n] = (sum_k X[m,k] * Wh[n,k] + bias[n]) * scale
// 128(M) x 64(N) tile, BK=32, 8 waves. Double-buffered async global->LDS
// staging (GLOBAL_LOAD_ASYNC_TO_LDS_B128 + s_wait_asynccnt).
// ---------------------------------------------------------------------------
#define GBM 128
#define GBN 64
#define GBK 32
#define GPAD 40   // halves per LDS row (80B -> conflict-free b128 row reads)

__global__ __launch_bounds__(256) void gemm_h(
    const _Float16* __restrict__ X, const _Float16* __restrict__ Wh,
    const float* __restrict__ bias,
    int M, int N, int Kdim, int ldx, int ldw,
    float* __restrict__ outF, _Float16* __restrict__ outH,
    _Float16* __restrict__ outT, int S, float scale) {
  __shared__ __align__(16) _Float16 sX[2][GBM * GPAD];
  __shared__ __align__(16) _Float16 sW[2][GBN * GPAD];

  const int tid  = threadIdx.x;
  const int wave = tid >> 5;
  const int lane = tid & 31;
  const int col  = lane & 15;
  const int hlf  = lane >> 4;
  const int kb   = hlf * 8;

  const int m0 = blockIdx.y * GBM;
  const int n0 = blockIdx.x * GBN;

  const int xrow = tid >> 1;        // 0..127, 2 threads/row (16 halves each)
  const int xcb  = (tid & 1) * 16;
  const int wrow = tid >> 2;        // 0..63, 4 threads/row (8 halves each)
  const int wcb  = (tid & 3) * 8;

  int wn = n0 + wrow; if (wn >= N) wn = 0;  // clamp (results guarded later)
  const _Float16* xbase = X + (size_t)(m0 + xrow) * ldx + xcb;
  const _Float16* wbase = Wh + (size_t)wn * ldw + wcb;
  const unsigned xo0 = (unsigned)(uintptr_t)&sX[0][xrow * GPAD + xcb];
  const unsigned xo1 = (unsigned)(uintptr_t)&sX[1][xrow * GPAD + xcb];
  const unsigned wo0 = (unsigned)(uintptr_t)&sW[0][wrow * GPAD + wcb];
  const unsigned wo1 = (unsigned)(uintptr_t)&sW[1][wrow * GPAD + wcb];

  v8f acc[4] = {};

  // prologue: stage k-slab 0 into buffer 0 (3 async b128 per thread)
  async_b128(xbase, xo0);
  async_b128(xbase + 8, xo0 + 16);
  async_b128(wbase, wo0);

  const int nsteps = Kdim / GBK;
  for (int s = 0; s < nsteps; ++s) {
    const int cur = s & 1;
    const bool more = (s + 1 < nsteps);
    if (more) {  // stage next slab into the other buffer
      const int kt = (s + 1) * GBK;
      async_b128(xbase + kt, cur ? xo0 : xo1);
      async_b128(xbase + kt + 8, (cur ? xo0 : xo1) + 16);
      async_b128(wbase + kt, cur ? wo0 : wo1);
      WAIT_ASYNC(3);  // current slab (3 older copies) complete
    } else {
      WAIT_ASYNC(0);
    }
    __syncthreads();

    v16h af = frag_from(&sX[cur][(wave * 16 + col) * GPAD], kb);
#pragma unroll
    for (int ns = 0; ns < 4; ++ns) {
      v16h bf = frag_from(&sW[cur][(ns * 16 + col) * GPAD], kb);
      acc[ns] = WMMA_F16(af, bf, acc[ns]);
    }
    __syncthreads();  // all waves done reading before buffer is re-staged
  }

  // ---- epilogue: bias, scale, store ----
#pragma unroll
  for (int ns = 0; ns < 4; ++ns) {
    const int n = n0 + ns * 16 + col;
    if (n >= N) continue;
    const float bv = bias[n];
#pragma unroll
    for (int r = 0; r < 8; ++r) {
      const int m = m0 + wave * 16 + r + hlf * 8;
      const float v = (acc[ns][r] + bv) * scale;
      if (outF) outF[(size_t)m * N + n] = v;
      if (outH) outH[(size_t)m * N + n] = (_Float16)v;
      if (outT) {
        const int bb = m / S, ss = m % S;
        outT[((size_t)bb * N + n) * S + ss] = (_Float16)v;
      }
    }
  }
}

// ---------------------------------------------------------------------------
// Fused attention (per wave: 16 query rows of one (b,h), two-pass online
// softmax). Pass A: scores = (Q/sqrt(D))K^T, masked, raw -> attnW, online
// (max,sum). Pass B: reload, normalize in place, P*V via LDS transpose.
// Requires D == 64 (true for all valid slim widths here).
// ---------------------------------------------------------------------------
__global__ __launch_bounds__(256) void attn_fused(
    const _Float16* __restrict__ Qh, const _Float16* __restrict__ Kh,
    const _Float16* __restrict__ Vt, const int* __restrict__ mask,
    float* __restrict__ attnW, _Float16* __restrict__ Oh,
    int H, int S, int width, int D) {
  if (D != 64) return;
  __shared__ __align__(16) _Float16 sP[8][16 * GPAD];

  const int tid  = threadIdx.x;
  const int wave = tid >> 5;
  const int lane = tid & 31;
  const int col  = lane & 15;
  const int hlf  = lane >> 4;
  const int kb   = hlf * 8;

  const int b  = blockIdx.z;
  const int h  = blockIdx.y;
  const int q0 = blockIdx.x * 128 + wave * 16;

  const _Float16* qp = Qh + ((size_t)(b * S + q0 + col)) * width + h * D;
  const v16h qf0 = frag_from(qp, kb);
  const v16h qf1 = frag_from(qp, 32 + kb);

  float mrow[8], srow[8];
#pragma unroll
  for (int r = 0; r < 8; ++r) { mrow[r] = -3.0e38f; srow[r] = 0.f; }

  float* wbase = attnW + (size_t)(b * H + h) * S * S;
  const int T = S >> 4;

  // -------- Pass A --------
  for (int t = 0; t < T; ++t) {
    const int krow = t * 16 + col;
    const _Float16* kp = Kh + ((size_t)(b * S + krow)) * width + h * D;
    v8f sc = {};
    sc = WMMA_F16(qf0, frag_from(kp, kb), sc);
    sc = WMMA_F16(qf1, frag_from(kp, 32 + kb), sc);
    const bool keep = mask[b * S + krow] != 0;
#pragma unroll
    for (int r = 0; r < 8; ++r) {
      const float x  = keep ? sc[r] : -1.0e9f;
      const float mn = fmaxf(mrow[r], x);
      srow[r] = srow[r] * __expf(mrow[r] - mn) + __expf(x - mn);
      mrow[r] = mn;
      wbase[(size_t)(q0 + r + hlf * 8) * S + krow] = x;
    }
  }

  // -------- cross-lane (m, s) merge within each 16-lane half --------
#pragma unroll
  for (int r = 0; r < 8; ++r) {
    float m = mrow[r], s = srow[r];
#pragma unroll
    for (int off = 1; off < 16; off <<= 1) {
      const float om = __shfl_xor(m, off, 32);
      const float os = __shfl_xor(s, off, 32);
      const float mn = fmaxf(m, om);
      s = s * __expf(m - mn) + os * __expf(om - mn);
      m = mn;
    }
    mrow[r] = m;
    srow[r] = 1.0f / s;
  }

  // -------- Pass B --------
  v8f oacc[4] = {};
  _Float16* myP = &sP[wave][0];
  const int T2 = S >> 5;
  for (int t2 = 0; t2 < T2; ++t2) {
#pragma unroll
    for (int sub = 0; sub < 2; ++sub) {
      const int krow = (t2 * 2 + sub) * 16 + col;
#pragma unroll
      for (int r = 0; r < 8; ++r) {
        const size_t idx = (size_t)(q0 + r + hlf * 8) * S + krow;
        const float p = __expf(wbase[idx] - mrow[r]) * srow[r];
        wbase[idx] = p;  // final normalized attention weight
        myP[(r + hlf * 8) * GPAD + sub * 16 + col] = (_Float16)p;
      }
    }
    wait_lds();  // same-wave LDS RAW before fragment reads

    const v16h pf = frag_from(&myP[col * GPAD], kb);
#pragma unroll
    for (int d16 = 0; d16 < 4; ++d16) {
      const _Float16* vp =
          Vt + ((size_t)b * width + h * D + d16 * 16 + col) * S + t2 * 32;
      oacc[d16] = WMMA_F16(pf, frag_from(vp, kb), oacc[d16]);
    }
  }

#pragma unroll
  for (int d16 = 0; d16 < 4; ++d16)
#pragma unroll
    for (int r = 0; r < 8; ++r)
      Oh[((size_t)(b * S + q0 + r + hlf * 8)) * width + h * D + d16 * 16 + col] =
          (_Float16)oacc[d16][r];
}

// ---------------------------------------------------------------------------
extern "C" void kernel_launch(void* const* d_in, const int* in_sizes, int n_in,
                              void* d_out, int out_size, void* d_ws,
                              size_t ws_size, hipStream_t stream) {
  const float* query = (const float*)d_in[0];
  const int*   mask  = (const int*)d_in[1];
  const float* q_w = (const float*)d_in[2];
  const float* q_b = (const float*)d_in[3];
  const float* k_w = (const float*)d_in[4];
  const float* k_b = (const float*)d_in[5];
  const float* v_w = (const float*)d_in[6];
  const float* v_b = (const float*)d_in[7];
  const float* o_w = (const float*)d_in[8];
  const float* o_b = (const float*)d_in[9];

  const int E = (int)(sqrt((double)in_sizes[2]) + 0.5);  // 1024
  const long long BS = in_sizes[1];                      // B*S

  // Recover width/H/S/B from out_size = BS*width + BS*H*S (unique for valid cfg)
  int width = 768, H = 12, S = 1024, Bn = (int)(BS / 1024);
  for (int w = 256; w <= 1024; ++w) {
    int h = (w * 16) / 1024; if (h < 1) h = 1;
    if (w % h) continue;
    if ((w / h) != 64) continue;               // kernels assume head_dim 64
    long long rem = (long long)out_size - BS * (long long)w;
    if (rem <= 0) continue;
    if (rem % (BS * (long long)h)) continue;
    long long s = rem / (BS * (long long)h);
    if (s <= 0 || (BS % s) || (s % 128)) continue;
    width = w; H = h; S = (int)s; Bn = (int)(BS / s);
    break;
  }
  const int D = width / H;
  const int M = (int)BS;

  // Workspace layout (all f16):
  //   Xh [M,width], Qh/Kh [M,width], Vt [B,width,S], AOh [M,width],
  //   Wqh/Wkh/Wvh/Woh [width,width]
  const size_t n1 = (size_t)M * width;
  const size_t nw = (size_t)width * width;
  _Float16* Xh  = (_Float16*)d_ws;
  _Float16* Qh  = Xh + n1;
  _Float16* Kh  = Qh + n1;
  _Float16* Vt  = Kh + n1;
  _Float16* AOh = Vt + n1;
  _Float16* Wqh = AOh + n1;
  _Float16* Wkh = Wqh + nw;
  _Float16* Wvh = Wkh + nw;
  _Float16* Woh = Wvh + nw;

  float* outO  = (float*)d_out;
  float* attnW = outO + (size_t)M * width;

  const dim3 blk(256);
  const float qscale = 1.0f / sqrtf((float)D);

  // One-time f32 -> f16 conversions (slim slices, packed ld = width).
  const dim3 gcx((width / 4 + 255) / 256, M);
  const dim3 gcw((width / 4 + 255) / 256, width);
  cvt_f16<<<gcx, blk, 0, stream>>>(query, Xh, width, E);
  cvt_f16<<<gcw, blk, 0, stream>>>(q_w, Wqh, width, E);
  cvt_f16<<<gcw, blk, 0, stream>>>(k_w, Wkh, width, E);
  cvt_f16<<<gcw, blk, 0, stream>>>(v_w, Wvh, width, E);
  cvt_f16<<<gcw, blk, 0, stream>>>(o_w, Woh, width, E);

  // Q/K/V projections; V stored transposed for the PV stage.
  const dim3 gp((width + GBN - 1) / GBN, (M + GBM - 1) / GBM);
  gemm_h<<<gp, blk, 0, stream>>>(Xh, Wqh, q_b, M, width, width, width, width,
                                 nullptr, Qh, nullptr, S, qscale);
  gemm_h<<<gp, blk, 0, stream>>>(Xh, Wkh, k_b, M, width, width, width, width,
                                 nullptr, Kh, nullptr, S, 1.0f);
  gemm_h<<<gp, blk, 0, stream>>>(Xh, Wvh, v_b, M, width, width, width, width,
                                 nullptr, nullptr, Vt, S, 1.0f);

  const dim3 ga(S / 128, H, Bn);
  attn_fused<<<ga, blk, 0, stream>>>(Qh, Kh, Vt, mask, attnW, AOh, H, S, width, D);

  // Output projection -> fp32 output region of d_out.
  gemm_h<<<gp, blk, 0, stream>>>(AOh, Woh, o_b, M, width, width, width, width,
                                 outO, nullptr, nullptr, S, 1.0f);
}
